// GRCN_88218628260836
// MI455X (gfx1250) — compile-verified, matrix operand order checked by reading, not run on hardware
//
#include <hip/hip_runtime.h>
#include <math.h>

#define NN 6144
#define FF 512
#define HH 128
#define CCH 16
#define KTOP 50

typedef __bf16 bf16;
typedef __attribute__((ext_vector_type(8)))  bf16  v8bf;
typedef __attribute__((ext_vector_type(16))) bf16  v16bf;
typedef __attribute__((ext_vector_type(8)))  float v8f;

// ---------------- reduction / elementwise kernels ----------------

__global__ void rowsum_dinv_kernel(const float* __restrict__ A, float* __restrict__ dinv, int n) {
    __shared__ float red[256];
    int row = blockIdx.x;
    const float* p = A + (size_t)row * n;
    float s = 0.f;
    for (int c = threadIdx.x; c < n; c += 256) s += p[c];
    red[threadIdx.x] = s;
    __syncthreads();
    for (int st = 128; st > 0; st >>= 1) {
        if (threadIdx.x < st) red[threadIdx.x] += red[threadIdx.x + st];
        __syncthreads();
    }
    if (threadIdx.x == 0) {
        float d = red[0];
        dinv[row] = d > 0.f ? 1.0f / sqrtf(d) : 0.f;
    }
}

// out[i,j] = bf16(dinv[i] * A[i,j] * dinv[j])
__global__ void symnorm_bf16_kernel(const float* __restrict__ A, const float* __restrict__ dinv,
                                    bf16* __restrict__ out, int n) {
    size_t idx = (size_t)blockIdx.x * 256 + threadIdx.x;
    size_t tot = (size_t)n * n;
    if (idx >= tot) return;
    int i = (int)(idx / n), j = (int)(idx % n);
    out[idx] = (bf16)(dinv[i] * A[idx] * dinv[j]);
}

// XT[c*rows + r] = bf16(X[r*cols + c] * w[c])   (column-scale + transpose)
__global__ void scale_transpose_bf16_kernel(const float* __restrict__ X, const float* __restrict__ w,
                                            bf16* __restrict__ XT, int rows, int cols) {
    size_t idx = (size_t)blockIdx.x * 256 + threadIdx.x;
    size_t tot = (size_t)rows * cols;
    if (idx >= tot) return;
    int c = (int)(idx / rows), r = (int)(idx % rows);
    XT[idx] = (bf16)(X[(size_t)r * cols + c] * w[c]);
}

// D[c*rows + r] = bf16(S[r*cols + c])
__global__ void transpose_bf16_kernel(const float* __restrict__ S, bf16* __restrict__ D,
                                      int rows, int cols) {
    size_t idx = (size_t)blockIdx.x * 256 + threadIdx.x;
    size_t tot = (size_t)rows * cols;
    if (idx >= tot) return;
    int c = (int)(idx / rows), r = (int)(idx % rows);
    D[idx] = (bf16)S[(size_t)r * cols + c];
}

__global__ void to_bf16_kernel(const float* __restrict__ S, bf16* __restrict__ D, size_t n) {
    size_t idx = (size_t)blockIdx.x * 256 + threadIdx.x;
    if (idx < n) D[idx] = (bf16)S[idx];
}

__global__ void zero_kernel(float* __restrict__ D, size_t n) {
    size_t idx = (size_t)blockIdx.x * 256 + threadIdx.x;
    if (idx < n) D[idx] = 0.f;
}

// row L2-normalize (clamped at 1e-12) and emit bf16
__global__ void rownorm_bf16_kernel(const float* __restrict__ E, bf16* __restrict__ EB, int cols) {
    __shared__ float red[256];
    __shared__ float s_scale;
    int row = blockIdx.x;
    const float* p = E + (size_t)row * cols;
    float s = 0.f;
    for (int c = threadIdx.x; c < cols; c += 256) { float v = p[c]; s += v * v; }
    red[threadIdx.x] = s;
    __syncthreads();
    for (int st = 128; st > 0; st >>= 1) {
        if (threadIdx.x < st) red[threadIdx.x] += red[threadIdx.x + st];
        __syncthreads();
    }
    if (threadIdx.x == 0) s_scale = 1.0f / fmaxf(sqrtf(red[0]), 1e-12f);
    __syncthreads();
    float sc = s_scale;
    bf16* q = EB + (size_t)row * cols;
    for (int c = threadIdx.x; c < cols; c += 256) q[c] = (bf16)(p[c] * sc);
}

// per-row top-K via K argmax passes over LDS copy; tie-break: lower index (lax.top_k)
__global__ void topk_kernel(const float* __restrict__ sim, float* __restrict__ vals,
                            int* __restrict__ idxs, int n, int k) {
    __shared__ float srow[NN];
    __shared__ float rv[256];
    __shared__ int   ri[256];
    int row = blockIdx.x;
    const float* p = sim + (size_t)row * n;
    for (int c = threadIdx.x; c < n; c += 256) srow[c] = p[c];
    __syncthreads();
    for (int t = 0; t < k; ++t) {
        float bv = -INFINITY; int bi = n;
        for (int c = threadIdx.x; c < n; c += 256) {
            float v = srow[c];
            if (v > bv || (v == bv && c < bi)) { bv = v; bi = c; }
        }
        rv[threadIdx.x] = bv; ri[threadIdx.x] = bi;
        __syncthreads();
        for (int st = 128; st > 0; st >>= 1) {
            if (threadIdx.x < st) {
                float ov = rv[threadIdx.x + st]; int oi = ri[threadIdx.x + st];
                if (ov > rv[threadIdx.x] || (ov == rv[threadIdx.x] && oi < ri[threadIdx.x])) {
                    rv[threadIdx.x] = ov; ri[threadIdx.x] = oi;
                }
            }
            __syncthreads();
        }
        if (threadIdx.x == 0) {
            vals[(size_t)row * k + t] = rv[0];
            idxs[(size_t)row * k + t] = ri[0];
            srow[ri[0]] = -INFINITY;
        }
        __syncthreads();
    }
}

// A_new[i,j] += 0.5*v ; A_new[j,i] += 0.5*v  (== (knn + knn^T)/2, deterministic: fp add commutes)
__global__ void scatter_kernel(const float* __restrict__ vals, const int* __restrict__ idxs,
                               float* __restrict__ Anew, int n, int k) {
    int t = blockIdx.x * 256 + threadIdx.x;
    if (t >= n * k) return;
    int i = t / k;
    int j = idxs[t];
    float v = 0.5f * vals[t];
    atomicAdd(&Anew[(size_t)i * n + j], v);
    atomicAdd(&Anew[(size_t)j * n + i], v);
}

// confidence mask (BETA = 0) + A_final = A_new_masked + Adj
__global__ void mask_final_kernel(float* __restrict__ Anew, float* __restrict__ Afin,
                                  const float* __restrict__ Adj, const float* __restrict__ u,
                                  const float* __restrict__ delta, int n) {
    size_t idx = (size_t)blockIdx.x * 256 + threadIdx.x;
    size_t tot = (size_t)n * n;
    if (idx >= tot) return;
    int i = (int)(idx / n), j = (int)(idx % n);
    float a = Anew[idx];
    float conf = a > 0.f ? expf(-u[j]) : 0.f;
    float weight = 2.f / (1.f + expf(delta[i] - conf));   // sigmoid(conf - delta)/0.5
    float mask = weight >= 1.f ? weight : 0.f;            // BETA == 0
    float am = a * mask;
    Anew[idx] = am;
    Afin[idx] = am + Adj[idx];
}

// ---------------- WMMA fragment helpers ----------------
// ISA 16-bit A/B 16x32 layout: lanes 0-15 hold K 0-7 & 16-23, lanes 16-31 hold K 8-15 & 24-31.
union frag16 { v16bf v; v8bf h[2]; };

__device__ __forceinline__ void load_frag(frag16& f, const bf16* p) {
    f.h[0] = *(const v8bf*)(p);
    f.h[1] = *(const v8bf*)(p + 16);
}

// ---------------- blocked WMMA GEMM (M%128==0, Ncols%128==0, K%32==0) ----------------
// C[M x Ncols] = A[M x K] * B, with B stored transposed BT[Ncols x K].
// 8 waves per block as 4(M) x 2(N); each wave owns a 32x64 register tile
// (2 A-frags x 4 B-frags -> 8 WMMAs per 12 b128 fragment loads: ~21 FLOP/B from L2).
__global__ void __launch_bounds__(256)
gemm_bf16_bt_big_kernel(const bf16* __restrict__ A, int lda,
                        const bf16* __restrict__ BT, int ldb,
                        float* __restrict__ C, int ldc,
                        int Kd, const float* __restrict__ bias, int relu, int transOut) {
    int wid  = threadIdx.x >> 5;
    int lane = threadIdx.x & 31;
    int half = lane >> 4;
    int r    = lane & 15;
    int tm0 = blockIdx.y * 8 + (wid >> 1) * 2;  // first of 2 m-tiles (16 rows each)
    int tn0 = blockIdx.x * 8 + (wid & 1) * 4;   // first of 4 n-tiles (16 cols each)

    const bf16* pa0 = A  + (size_t)(tm0 * 16 + r) * lda + half * 8;
    const bf16* pa1 = pa0 + (size_t)16 * lda;
    const bf16* pb0 = BT + (size_t)(tn0 * 16 + r) * ldb + half * 8;
    const bf16* pb1 = pb0 + (size_t)16 * ldb;
    const bf16* pb2 = pb1 + (size_t)16 * ldb;
    const bf16* pb3 = pb2 + (size_t)16 * ldb;

    v8f acc[2][4];
    #pragma unroll
    for (int i = 0; i < 2; ++i)
        #pragma unroll
        for (int j = 0; j < 4; ++j) acc[i][j] = (v8f){};

    for (int k0 = 0; k0 < Kd; k0 += 32) {
        frag16 a0, a1, b0, b1, b2, b3;
        load_frag(a0, pa0); load_frag(a1, pa1);
        load_frag(b0, pb0); load_frag(b1, pb1);
        load_frag(b2, pb2); load_frag(b3, pb3);
        acc[0][0] = __builtin_amdgcn_wmma_f32_16x16x32_bf16(false, a0.v, false, b0.v, (short)0, acc[0][0], false, false);
        acc[0][1] = __builtin_amdgcn_wmma_f32_16x16x32_bf16(false, a0.v, false, b1.v, (short)0, acc[0][1], false, false);
        acc[0][2] = __builtin_amdgcn_wmma_f32_16x16x32_bf16(false, a0.v, false, b2.v, (short)0, acc[0][2], false, false);
        acc[0][3] = __builtin_amdgcn_wmma_f32_16x16x32_bf16(false, a0.v, false, b3.v, (short)0, acc[0][3], false, false);
        acc[1][0] = __builtin_amdgcn_wmma_f32_16x16x32_bf16(false, a1.v, false, b0.v, (short)0, acc[1][0], false, false);
        acc[1][1] = __builtin_amdgcn_wmma_f32_16x16x32_bf16(false, a1.v, false, b1.v, (short)0, acc[1][1], false, false);
        acc[1][2] = __builtin_amdgcn_wmma_f32_16x16x32_bf16(false, a1.v, false, b2.v, (short)0, acc[1][2], false, false);
        acc[1][3] = __builtin_amdgcn_wmma_f32_16x16x32_bf16(false, a1.v, false, b3.v, (short)0, acc[1][3], false, false);
        pa0 += 32; pa1 += 32; pb0 += 32; pb1 += 32; pb2 += 32; pb3 += 32;
    }

    #pragma unroll
    for (int j = 0; j < 4; ++j) {
        int ncol = (tn0 + j) * 16 + r;
        float bv = bias ? bias[ncol] : 0.f;
        #pragma unroll
        for (int i = 0; i < 2; ++i) {
            #pragma unroll
            for (int v = 0; v < 8; ++v) {
                int mrow = (tm0 + i) * 16 + half * 8 + v;
                float x = acc[i][j][v] + bv;
                if (relu) x = fmaxf(x, 0.f);
                size_t o = transOut ? (size_t)ncol * ldc + mrow : (size_t)mrow * ldc + ncol;
                C[o] = x;
            }
        }
    }
}

// ---------------- simple WMMA GEMM (wave per 16x16 tile) for small Ncols ----------------
__global__ void __launch_bounds__(256)
gemm_bf16_bt_kernel(const bf16* __restrict__ A, int lda,
                    const bf16* __restrict__ BT, int ldb,
                    float* __restrict__ C, int ldc,
                    int M, int Ncols, int Kd,
                    const float* __restrict__ bias, int relu, int transOut) {
    int wid  = threadIdx.x >> 5;
    int lane = threadIdx.x & 31;
    int half = lane >> 4;
    int r    = lane & 15;
    int tilesM = M >> 4, tilesN = Ncols >> 4;
    int tm = blockIdx.y * 2 + (wid >> 2);
    int tn = blockIdx.x * 4 + (wid & 3);
    if (tm >= tilesM || tn >= tilesN) return;

    const bf16* pa = A  + (size_t)(tm * 16 + r) * lda + half * 8;
    const bf16* pb = BT + (size_t)(tn * 16 + r) * ldb + half * 8;

    v8f acc = {};
    for (int k0 = 0; k0 < Kd; k0 += 32) {
        frag16 ua, ub;
        load_frag(ua, pa);
        load_frag(ub, pb);
        acc = __builtin_amdgcn_wmma_f32_16x16x32_bf16(
            false, ua.v, false, ub.v, (short)0, acc, false, false);
        pa += 32; pb += 32;
    }

    int ncol = tn * 16 + r;
    float bv = bias ? bias[ncol] : 0.f;
    #pragma unroll
    for (int v = 0; v < 8; ++v) {
        int mrow = tm * 16 + half * 8 + v;
        float x = acc[v] + bv;
        if (relu) x = fmaxf(x, 0.f);
        size_t o = transOut ? (size_t)ncol * ldc + mrow : (size_t)mrow * ldc + ncol;
        C[o] = x;
    }
}

static inline dim3 gemm_grid_small(int M, int Ncols) {
    int tilesM = M / 16, tilesN = Ncols / 16;
    return dim3((unsigned)((tilesN + 3) / 4), (unsigned)((tilesM + 1) / 2));
}
static inline dim3 gemm_grid_big(int M, int Ncols) {
    return dim3((unsigned)(Ncols / 128), (unsigned)(M / 128));
}
static inline dim3 g1(size_t n) { return dim3((unsigned)((n + 255) / 256)); }

// ---------------- launch ----------------

extern "C" void kernel_launch(void* const* d_in, const int* in_sizes, int n_in,
                              void* d_out, int out_size, void* d_ws, size_t ws_size,
                              hipStream_t stream) {
    const int N = NN, F = FF, H = HH, C = CCH, K = KTOP;
    const float* input = (const float*)d_in[0];
    const float* Adj   = (const float*)d_in[1];
    const float* unc   = (const float*)d_in[2];
    const float* w1    = (const float*)d_in[3];
    const float* w2    = (const float*)d_in[4];
    const float* delta = (const float*)d_in[5];
    const float* W1    = (const float*)d_in[6];
    const float* b1    = (const float*)d_in[7];
    const float* W2    = (const float*)d_in[8];
    const float* b2    = (const float*)d_in[9];
    (void)in_sizes; (void)n_in; (void)out_size; (void)ws_size;

    float* outX = (float*)d_out;                 // [N, C]
    float* Anew = outX + (size_t)N * C;          // [N, N]
    float* Afin = Anew + (size_t)N * N;          // [N, N]

    char* p = (char*)d_ws;
    auto take = [&](size_t b) -> void* {
        void* r = (void*)p;
        p += (b + 255) & ~(size_t)255;
        return r;
    };
    float* dinv  = (float*)take((size_t)N * 4);
    float* dinv2 = (float*)take((size_t)N * 4);
    bf16*  Anb   = (bf16*) take((size_t)N * N * 2);   // An, later reused as Afn
    float* sim   = (float*)take((size_t)N * N * 4);
    bf16*  embB  = (bf16*) take((size_t)N * F * 2);
    char*  r1    = (char*) take((size_t)N * F * 4);   // phase1: h   | phase2: XB, W1T, W2T
    char*  r2    = (char*) take((size_t)N * F * 4);   // phase1: emb | phase2: PT,PTB,Q,QB,RT,RTB
    char*  r3    = (char*) take((size_t)F * N * 2);   // phase1: T1  | phase2: tkv, tki

    // phase-1 aliases
    float* h   = (float*)r1;
    float* emb = (float*)r2;
    bf16*  T1  = (bf16*)r3;
    // phase-2 aliases
    float* tkv = (float*)r3;
    int*   tki = (int*)(r3 + (((size_t)N * K * 4 + 255) & ~(size_t)255));
    bf16*  XB  = (bf16*)r1;
    bf16*  W1T = (bf16*)(r1 + (((size_t)N * F * 2 + 255) & ~(size_t)255));
    bf16*  W2T = (bf16*)((char*)W1T + (((size_t)H * F * 2 + 255) & ~(size_t)255));
    float* PT  = (float*)r2;
    bf16*  PTB = (bf16*)(r2 + (((size_t)H * N * 4 + 255) & ~(size_t)255));
    float* Q   = (float*)((char*)PTB + (((size_t)H * N * 2 + 255) & ~(size_t)255));
    bf16*  QB  = (bf16*)((char*)Q + (((size_t)N * H * 4 + 255) & ~(size_t)255));
    float* RT  = (float*)((char*)QB + (((size_t)N * H * 2 + 255) & ~(size_t)255));
    bf16*  RTB = (bf16*)((char*)RT + (((size_t)C * N * 4 + 255) & ~(size_t)255));

    dim3 blk(256);

    // An = D^-1/2 Adj D^-1/2 (bf16)
    rowsum_dinv_kernel<<<N, blk, 0, stream>>>(Adj, dinv, N);
    symnorm_bf16_kernel<<<g1((size_t)N * N), blk, 0, stream>>>(Adj, dinv, Anb, N);

    // h = An @ (input * w1)          [M=N, Ncols=F, K=N] -> blocked
    scale_transpose_bf16_kernel<<<g1((size_t)N * F), blk, 0, stream>>>(input, w1, T1, N, F);
    gemm_bf16_bt_big_kernel<<<gemm_grid_big(N, F), blk, 0, stream>>>(Anb, N, T1, N, h, F, N, nullptr, 0, 0);

    // emb = An @ (h * w2)            [M=N, Ncols=F, K=N] -> blocked
    scale_transpose_bf16_kernel<<<g1((size_t)N * F), blk, 0, stream>>>(h, w2, T1, N, F);
    gemm_bf16_bt_big_kernel<<<gemm_grid_big(N, F), blk, 0, stream>>>(Anb, N, T1, N, emb, F, N, nullptr, 0, 0);

    // row-normalize -> embB ; sim = embB @ embB^T   [M=N, Ncols=N, K=F] -> blocked
    rownorm_bf16_kernel<<<N, blk, 0, stream>>>(emb, embB, F);
    gemm_bf16_bt_big_kernel<<<gemm_grid_big(N, N), blk, 0, stream>>>(embB, F, embB, F, sim, N, F, nullptr, 0, 0);

    // top-K per row, symmetric scatter, confidence mask, A_final
    topk_kernel<<<N, blk, 0, stream>>>(sim, tkv, tki, N, K);
    zero_kernel<<<g1((size_t)N * N), blk, 0, stream>>>(Anew, (size_t)N * N);
    scatter_kernel<<<g1((size_t)N * K), blk, 0, stream>>>(tkv, tki, Anew, N, K);
    mask_final_kernel<<<g1((size_t)N * N), blk, 0, stream>>>(Anew, Afin, Adj, unc, delta, N);

    // Afn = D^-1/2 A_final D^-1/2 (reuse Anb)
    rowsum_dinv_kernel<<<N, blk, 0, stream>>>(Afin, dinv2, N);
    symnorm_bf16_kernel<<<g1((size_t)N * N), blk, 0, stream>>>(Afin, dinv2, Anb, N);

    // task GCN: x = Afn @ relu(Afn @ (X W1) + b1) W2 + b2
    to_bf16_kernel<<<g1((size_t)N * F), blk, 0, stream>>>(input, XB, (size_t)N * F);
    transpose_bf16_kernel<<<g1((size_t)F * H), blk, 0, stream>>>(W1, W1T, F, H);
    // P^T = (X @ W1)^T               [M=N, Ncols=H=128, K=F] -> blocked, transposed out
    gemm_bf16_bt_big_kernel<<<gemm_grid_big(N, H), blk, 0, stream>>>(XB, F, W1T, F, PT, N, F, nullptr, 0, 1);
    to_bf16_kernel<<<g1((size_t)H * N), blk, 0, stream>>>(PT, PTB, (size_t)H * N);
    // Q = relu(Afn @ P + b1)         [M=N, Ncols=H, K=N] -> blocked
    gemm_bf16_bt_big_kernel<<<gemm_grid_big(N, H), blk, 0, stream>>>(Anb, N, PTB, N, Q, H, N, b1, 1, 0);
    to_bf16_kernel<<<g1((size_t)N * H), blk, 0, stream>>>(Q, QB, (size_t)N * H);
    transpose_bf16_kernel<<<g1((size_t)H * C), blk, 0, stream>>>(W2, W2T, H, C);
    // R^T = (Q @ W2)^T               [Ncols=16] -> simple kernel
    gemm_bf16_bt_kernel<<<gemm_grid_small(N, C), blk, 0, stream>>>(QB, H, W2T, H, RT, N, N, C, H, nullptr, 0, 1);
    to_bf16_kernel<<<g1((size_t)C * N), blk, 0, stream>>>(RT, RTB, (size_t)C * N);
    // x = Afn @ R + b2               [Ncols=16] -> simple kernel
    gemm_bf16_bt_kernel<<<gemm_grid_small(N, C), blk, 0, stream>>>(Anb, N, RTB, N, outX, C, N, C, N, b2, 0, 0);
}
